// AttentionSTAE_12627203850706
// MI455X (gfx1250) — compile-verified, matrix operand order checked by reading, not run on hardware
//
#include <hip/hip_runtime.h>
#include <hip/hip_bf16.h>

// ---------------------------------------------------------------------------
// STAE pipeline for MI455X (gfx1250, wave32, WMMA).
//   - dense GEMMs: v_wmma_f32_16x16x32_f16, one 16x16 C-tile per wave,
//     branch-free loaders (global_load_b128 + v_cvt_pk_rtz_f16_f32)
//   - fused LSTM: Wih/Whh staged once into LDS as f16 (up to 300KB of the
//     320KB/WGP budget); per-step A/B fragments come from LDS (ds_load_b32)
//   - GAT softmax-aggregation with on-the-fly edge generation + float atomics
// Assumptions: all GEMMs K%32==0 (hin zero-padded 26->32, Wih0 zero-padded in
// the LDS staging loop), M%16==0; only N may be ragged (decoder gates N=40).
// ---------------------------------------------------------------------------

typedef __attribute__((ext_vector_type(16))) _Float16 v16h;
typedef __attribute__((ext_vector_type(8)))  float    v8f;
typedef __attribute__((ext_vector_type(2)))  __fp16   h2raw;

union Frag {
    v16h     v;
    h2raw    p[8];
    unsigned u[8];
};

// problem constants (from setup_inputs)
#define CN    64      // turbines
#define CB    64      // batch
#define CT    48      // time steps
#define CF    10      // features
#define CEMB  16
#define CH    128     // LSTM hidden
#define CHD1  64
#define CHD2  32
#define CE    256     // edges per graph
#define CK0   32      // hin feature dim padded 26 -> 32 for branch-free WMMA
#define CNB   (CN * CB)                 // 4096 sequences
#define CM    (CB * CT)                 // 3072 replicated graphs
#define CNN   ((long)CM * CN)           // 196608 graph nodes
#define CME   ((long)CM * CE)           // 786432 non-loop edges
#define CEMT  (CME + CNN)               // 983040 total edges (with self loops)

__device__ __forceinline__ float dsigm(float x) { return 1.f / (1.f + __expf(-x)); }

// monotonic float<->uint encoding so atomicMax(unsigned) == float max
__device__ __forceinline__ unsigned fenc(float f) {
    unsigned u = __float_as_uint(f);
    return (u & 0x80000000u) ? ~u : (u | 0x80000000u);
}
__device__ __forceinline__ float fdec(unsigned u) {
    return (u & 0x80000000u) ? __uint_as_float(u & 0x7FFFFFFFu) : __uint_as_float(~u);
}

// K-offset pattern for 16-bit A/B fragments (ISA 7.12.2), pairs of consecutive K
__device__ __forceinline__ int kpat(int v, int hi) {
    return (v < 4) ? (hi * 8 + v * 2) : (16 + hi * 8 + (v - 4) * 2);
}

// ---------------- branch-free WMMA fragment loaders ---------------------------
// A fragment 16x32 (MxK) f16 from global f32 row-major [.., lda]
__device__ __forceinline__ v16h load_a_fast(const float* __restrict__ A, long lda,
                                            long row0, int k0) {
    int lane = threadIdx.x & 31;
    int m = lane & 15, hi = lane >> 4;
    const float* Ar = A + (row0 + m) * lda + k0;
    Frag f;
#pragma unroll
    for (int v = 0; v < 8; ++v) {
        float2 q = *(const float2*)(Ar + kpat(v, hi));
        f.p[v] = __builtin_amdgcn_cvt_pkrtz(q.x, q.y);
    }
    return f.v;
}

// B fragment (32x16 KxN) from weight W[N,K] row-major (B[k][n] = W[n][k])
__device__ __forceinline__ v16h load_b_fast(const float* __restrict__ W, int ldw,
                                            int col0, int k0) {
    int lane = threadIdx.x & 31;
    int n = lane & 15, hi = lane >> 4;
    const float* Wr = W + (long)(col0 + n) * ldw + k0;
    Frag f;
#pragma unroll
    for (int v = 0; v < 8; ++v) {
        float2 q = *(const float2*)(Wr + kpat(v, hi));
        f.p[v] = __builtin_amdgcn_cvt_pkrtz(q.x, q.y);
    }
    return f.v;
}

// B fragment with a single hoisted column predicate (ragged N, K still %32==0)
__device__ __forceinline__ v16h load_b_guardN(const float* __restrict__ W, int ldw,
                                              int col0, int k0, int Nmax) {
    int lane = threadIdx.x & 31;
    int n = lane & 15, hi = lane >> 4;
    int c = col0 + n;
    Frag f;
    if (c < Nmax) {
        const float* Wr = W + (long)c * ldw + k0;
#pragma unroll
        for (int v = 0; v < 8; ++v) {
            float2 q = *(const float2*)(Wr + kpat(v, hi));
            f.p[v] = __builtin_amdgcn_cvt_pkrtz(q.x, q.y);
        }
    } else {
#pragma unroll
        for (int v = 0; v < 8; ++v) f.u[v] = 0u;
    }
    return f.v;
}

// A fragment from LDS f16 tile [16][ldh]: 8 packed 32-bit ds loads
__device__ __forceinline__ v16h load_a_lds(const _Float16* __restrict__ Hs, int ldh, int k0) {
    int lane = threadIdx.x & 31;
    int m = lane & 15, hi = lane >> 4;
    const _Float16* Hr = Hs + m * ldh + k0;
    Frag f;
#pragma unroll
    for (int v = 0; v < 8; ++v)
        f.u[v] = *(const unsigned*)(Hr + kpat(v, hi));
    return f.v;
}

// B fragment from LDS f16 weight tile W[N][ldw]: 8 packed 32-bit ds loads
__device__ __forceinline__ v16h load_b_lds(const _Float16* __restrict__ Ws, int ldw,
                                           int col0, int k0) {
    int lane = threadIdx.x & 31;
    int n = lane & 15, hi = lane >> 4;
    const _Float16* Wr = Ws + (long)(col0 + n) * ldw + k0;
    Frag f;
#pragma unroll
    for (int v = 0; v < 8; ++v)
        f.u[v] = *(const unsigned*)(Wr + kpat(v, hi));
    return f.v;
}

// ---------------- generic GEMM: C[M,N] = act(A[M,K] @ W[N,K]^T + b1 + b2) ----
// act: 0 none, 1 relu, 2 sigmoid.   Requires K%32==0, M%16==0.
__global__ void gemm_wt_kernel(const float* __restrict__ A, const float* __restrict__ W,
                               const float* __restrict__ b1, const float* __restrict__ b2,
                               float* __restrict__ C, long M, int N, int K, int act) {
    int wave = threadIdx.x >> 5;
    long tile = (long)blockIdx.x * 8 + wave;
    int ntn = (N + 15) >> 4;
    long ntiles = (M >> 4) * (long)ntn;
    if (tile >= ntiles) return;                 // wave-uniform
    long mt = tile / ntn;
    int  nt = (int)(tile % ntn);
    long row0 = mt << 4;
    int  col0 = nt << 4;

    v8f acc = {};
    if (col0 + 16 <= N) {                       // interior tile: zero predication
#pragma unroll 2
        for (int k0 = 0; k0 < K; k0 += 32) {
            __builtin_prefetch(W + (long)col0 * K + k0 + 64, 0, 1);  // global_prefetch_b8
            v16h a = load_a_fast(A, K, row0, k0);
            v16h b = load_b_fast(W, K, col0, k0);
            acc = __builtin_amdgcn_wmma_f32_16x16x32_f16(false, a, false, b,
                                                         (short)0, acc, false, false);
        }
    } else {                                    // ragged N (decoder gates N=40)
#pragma unroll 2
        for (int k0 = 0; k0 < K; k0 += 32) {
            v16h a = load_a_fast(A, K, row0, k0);
            v16h b = load_b_guardN(W, K, col0, k0, N);
            acc = __builtin_amdgcn_wmma_f32_16x16x32_f16(false, a, false, b,
                                                         (short)0, acc, false, false);
        }
    }
    int lane = threadIdx.x & 31;
    int n = lane & 15, hi = lane >> 4;
    int c = col0 + n;
    if (c < N) {
        float bv = (b1 ? b1[c] : 0.f) + (b2 ? b2[c] : 0.f);
#pragma unroll
        for (int v = 0; v < 8; ++v) {
            long r = row0 + v + 8 * hi;
            float x = acc[v] + bv;
            if (act == 1)      x = fmaxf(x, 0.f);
            else if (act == 2) x = dsigm(x);
            C[r * N + c] = x;
        }
    }
}

// ---------------- fused LSTM layer (H%16==0, Kin%32==0, NB%16==0) ------------
// block = 16 batch rows, 8 waves cover the 4H/16 gate tiles each time step.
// Wih ([4H,KinSrc], zero-padded to Kin) and Whh ([4H,H]) are staged once into
// LDS as f16; the 48-step recurrence then reads all B fragments from LDS.
__global__ void lstm_rec_kernel(const float* __restrict__ xin, const float* __restrict__ Wih,
                                const float* __restrict__ Whh, const float* __restrict__ bih,
                                const float* __restrict__ bhh, float* __restrict__ hs,
                                int NB, int T, int Kin, int KinSrc, int H) {
    extern __shared__ char smem[];
    const int G4 = 4 * H;
    _Float16* h_sh   = (_Float16*)smem;                    // [16][H]
    float*    c_sh   = (float*)(h_sh + 16 * H);            // [16][H]
    float*    g_sh   = c_sh + 16 * H;                      // [16][4H]
    _Float16* wih_sh = (_Float16*)(g_sh + 16 * G4);        // [4H][Kin]  (zero-padded)
    _Float16* whh_sh = wih_sh + (long)G4 * Kin;            // [4H][H]
    long row0 = (long)blockIdx.x * 16;
    int tid = threadIdx.x, wave = tid >> 5, lane = tid & 31;

    for (int i = tid; i < 16 * H; i += blockDim.x) { h_sh[i] = (_Float16)0.f; c_sh[i] = 0.f; }
    for (long i = tid; i < (long)G4 * Kin; i += blockDim.x) {   // stage Wih (pad K)
        int n = (int)(i / Kin), k = (int)(i % Kin);
        wih_sh[i] = (_Float16)((k < KinSrc) ? Wih[(long)n * KinSrc + k] : 0.f);
    }
    for (long i = tid; i < (long)G4 * H; i += blockDim.x)       // stage Whh
        whh_sh[i] = (_Float16)Whh[i];
    __syncthreads();

    int ntn = G4 >> 4;
    int tpw = (ntn + 7) >> 3;
    for (int t = 0; t < T; ++t) {
        const float* At = xin + (long)t * Kin;             // row stride = T*Kin
        for (int w = 0; w < tpw; ++w) {
            int nt = wave * tpw + w;
            if (nt < ntn) {                                // wave-uniform
                int col0 = nt << 4;
                v8f acc = {};
#pragma unroll 2
                for (int k0 = 0; k0 < Kin; k0 += 32) {     // x_t @ Wih^T (x global, W LDS)
                    v16h a = load_a_fast(At, (long)T * Kin, row0, k0);
                    v16h b = load_b_lds(wih_sh, Kin, col0, k0);
                    acc = __builtin_amdgcn_wmma_f32_16x16x32_f16(false, a, false, b,
                                                                 (short)0, acc, false, false);
                }
#pragma unroll 2
                for (int k0 = 0; k0 < H; k0 += 32) {       // h_{t-1} @ Whh^T (all LDS)
                    v16h a = load_a_lds(h_sh, H, k0);
                    v16h b = load_b_lds(whh_sh, H, col0, k0);
                    acc = __builtin_amdgcn_wmma_f32_16x16x32_f16(false, a, false, b,
                                                                 (short)0, acc, false, false);
                }
                int n = lane & 15, hi = lane >> 4;
                int c = col0 + n;
                float bv = bih[c] + bhh[c];
#pragma unroll
                for (int v = 0; v < 8; ++v)
                    g_sh[(v + 8 * hi) * G4 + c] = acc[v] + bv;
            }
        }
        __syncthreads();
        for (int i = tid; i < 16 * H; i += blockDim.x) {   // gate math (i,f,g,o)
            int m = i / H, j = i - m * H;
            float gi = g_sh[m * G4 + j];
            float gf = g_sh[m * G4 + H + j];
            float gg = g_sh[m * G4 + 2 * H + j];
            float go = g_sh[m * G4 + 3 * H + j];
            float cc = dsigm(gf) * c_sh[i] + dsigm(gi) * tanhf(gg);
            float hh = dsigm(go) * tanhf(cc);
            c_sh[i] = cc;
            h_sh[i] = (_Float16)hh;
            hs[((row0 + m) * T + t) * (long)H + j] = hh;
        }
        __syncthreads();
    }
}

// ---------------- elementwise / layout kernels --------------------------------
__global__ void fill_f32(float* p, float v, long n) {
    long i = (long)blockIdx.x * blockDim.x + threadIdx.x;
    if (i < n) p[i] = v;
}
__global__ void fill_u32(unsigned* p, unsigned v, long n) {
    long i = (long)blockIdx.x * blockDim.x + threadIdx.x;
    if (i < n) p[i] = v;
}

// hin[(n*B+b), t, k] = k<F ? x[n,b,t,k] : k<F+EMB ? emb[n,k-F] : 0  (K padded to 32)
__global__ void build_hin(const float* __restrict__ x, const float* __restrict__ emb,
                          float* __restrict__ hin) {
    long total = (long)CNB * CT * CK0;
    long i = (long)blockIdx.x * blockDim.x + threadIdx.x;
    if (i >= total) return;
    int k = (int)(i % CK0);
    long rt = i / CK0;
    int t = (int)(rt % CT);
    long r = rt / CT;                 // n*B+b
    int n = (int)(r / CB);
    float v = 0.f;
    if (k < CF)             v = x[(r * CT + t) * CF + k];
    else if (k < CF + CEMB) v = emb[n * CEMB + (k - CF)];
    hin[i] = v;
}

// th[(n*B+b), t, H] -> xf[(b*T+t)*N + n, H]
__global__ void th_to_xf(const float* __restrict__ th, float* __restrict__ xf) {
    long total = CNN * CH;
    long o = (long)blockIdx.x * blockDim.x + threadIdx.x;
    if (o >= total) return;
    int j = (int)(o % CH);
    long node = o / CH;
    int n = (int)(node % CN);
    long mr = node / CN;              // b*T+t
    int t = (int)(mr % CT);
    long b = mr / CT;
    xf[o] = th[(((long)n * CB + b) * CT + t) * CH + j];
}

// y[(b*T+t)*N + n, H] -> yd[(n*B+b)*T + t, H]
__global__ void y_to_yd(const float* __restrict__ y, float* __restrict__ yd) {
    long total = CNN * CH;
    long o = (long)blockIdx.x * blockDim.x + threadIdx.x;
    if (o >= total) return;
    int j = (int)(o % CH);
    long q = o / CH;
    int t = (int)(q % CT);
    long r = q / CT;                  // n*B+b
    long n = r / CB, b = r % CB;
    yd[o] = y[(((long)b * CT + t) * CN + n) * CH + j];
}

__global__ void bias_relu(float* __restrict__ p, const float* __restrict__ bias, long rows, int Nf) {
    long i = (long)blockIdx.x * blockDim.x + threadIdx.x;
    if (i >= rows * Nf) return;
    float v = p[i] + bias[i % Nf];
    p[i] = fmaxf(v, 0.f);
}

// ---------------- GAT pieces --------------------------------------------------
__device__ __forceinline__ void edge_sd(long e, const int* adj, long& s, long& d) {
    if (e < CME) {
        long g = e / CE;
        int  j = (int)(e % CE);
        s = g * CN + adj[j];
        d = g * CN + adj[CE + j];
    } else {
        s = d = e - CME;
    }
}

__global__ void gat_alpha(const float* __restrict__ h, const float* __restrict__ a_s,
                          const float* __restrict__ a_d, float* __restrict__ as_,
                          float* __restrict__ ad_, int heads, int od) {
    long i = (long)blockIdx.x * blockDim.x + threadIdx.x;
    if (i >= CNN * heads) return;
    long node = i / heads;
    int hd = (int)(i % heads);
    const float* hp = h + (node * heads + hd) * (long)od;
    float ss = 0.f, sd = 0.f;
    for (int j = 0; j < od; ++j) { ss += hp[j] * a_s[hd * od + j]; sd += hp[j] * a_d[hd * od + j]; }
    as_[i] = ss; ad_[i] = sd;
}

__global__ void gat_edge_max(const float* __restrict__ as_, const float* __restrict__ ad_,
                             unsigned* __restrict__ m_u, const int* __restrict__ adj, int heads) {
    long i = (long)blockIdx.x * blockDim.x + threadIdx.x;
    if (i >= CEMT * heads) return;
    long e = i / heads;
    int hd = (int)(i % heads);
    long s, d; edge_sd(e, adj, s, d);
    float ev = as_[s * heads + hd] + ad_[d * heads + hd];
    ev = ev > 0.f ? ev : 0.2f * ev;                     // leaky_relu
    atomicMax(&m_u[d * heads + hd], fenc(ev));
}

__global__ void gat_edge_exp(const float* __restrict__ as_, const float* __restrict__ ad_,
                             const unsigned* __restrict__ m_u, float* __restrict__ ex,
                             float* __restrict__ denom, const int* __restrict__ adj, int heads) {
    long i = (long)blockIdx.x * blockDim.x + threadIdx.x;
    if (i >= CEMT * heads) return;
    long e = i / heads;
    int hd = (int)(i % heads);
    long s, d; edge_sd(e, adj, s, d);
    float ev = as_[s * heads + hd] + ad_[d * heads + hd];
    ev = ev > 0.f ? ev : 0.2f * ev;
    float x = __expf(ev - fdec(m_u[d * heads + hd]));
    ex[i] = x;
    atomicAdd(&denom[d * heads + hd], x);
}

__global__ void gat_aggregate(const float* __restrict__ hfeat, const float* __restrict__ ex,
                              const float* __restrict__ denom, const int* __restrict__ adj,
                              float* __restrict__ out, int heads, int od) {
    long e = (long)blockIdx.x * blockDim.x + threadIdx.x;
    if (e >= CEMT) return;
    long s, d; edge_sd(e, adj, s, d);
    for (int hd = 0; hd < heads; ++hd) {
        float a = ex[e * heads + hd] / denom[d * heads + hd];
        const float* hp = hfeat + (s * heads + hd) * (long)od;
        float* op = out + (d * heads + hd) * (long)od;
        for (int j = 0; j < od; ++j) atomicAdd(&op[j], a * hp[j]);
    }
}

// ---------------- tiny decoder LSTM (H=10) -----------------------------------
__global__ void lstm_small_l0(const float* __restrict__ gates, const float* __restrict__ Whh,
                              float* __restrict__ hs) {
    int r = blockIdx.x * blockDim.x + threadIdx.x;
    if (r >= CNB) return;
    float h[CF], c[CF];
    for (int j = 0; j < CF; ++j) { h[j] = 0.f; c[j] = 0.f; }
    for (int t = 0; t < CT; ++t) {
        const float* g0 = gates + ((long)r * CT + t) * (4 * CF);
        float g[4 * CF];
        for (int gi = 0; gi < 4 * CF; ++gi) {
            float s = g0[gi];
            for (int j = 0; j < CF; ++j) s += h[j] * Whh[gi * CF + j];
            g[gi] = s;
        }
        for (int j = 0; j < CF; ++j) {
            c[j] = dsigm(g[CF + j]) * c[j] + dsigm(g[j]) * tanhf(g[2 * CF + j]);
            h[j] = dsigm(g[3 * CF + j]) * tanhf(c[j]);
            hs[((long)r * CT + t) * CF + j] = h[j];
        }
    }
}

__global__ void lstm_small_l1_sig(const float* __restrict__ xin, const float* __restrict__ Wih,
                                  const float* __restrict__ Whh, const float* __restrict__ bih,
                                  const float* __restrict__ bhh, float* __restrict__ out) {
    int r = blockIdx.x * blockDim.x + threadIdx.x;
    if (r >= CNB) return;
    float h[CF], c[CF];
    for (int j = 0; j < CF; ++j) { h[j] = 0.f; c[j] = 0.f; }
    for (int t = 0; t < CT; ++t) {
        const float* xt = xin + ((long)r * CT + t) * CF;
        float g[4 * CF];
        for (int gi = 0; gi < 4 * CF; ++gi) {
            float s = bih[gi] + bhh[gi];
            for (int j = 0; j < CF; ++j) s += xt[j] * Wih[gi * CF + j] + h[j] * Whh[gi * CF + j];
            g[gi] = s;
        }
        for (int j = 0; j < CF; ++j) {
            c[j] = dsigm(g[CF + j]) * c[j] + dsigm(g[j]) * tanhf(g[2 * CF + j]);
            h[j] = dsigm(g[3 * CF + j]) * tanhf(c[j]);
            out[((long)r * CT + t) * CF + j] = dsigm(h[j]);   // output = sigmoid(hs)
        }
    }
}

// ---------------- host-side helpers ------------------------------------------
static inline long cdivl(long a, long b) { return (a + b - 1) / b; }

static void run_gemm(hipStream_t s, const float* A, const float* W, const float* b1,
                     const float* b2, float* C, long M, int N, int K, int act) {
    long tiles = (M >> 4) * cdivl(N, 16);
    long blocks = cdivl(tiles, 8);
    gemm_wt_kernel<<<dim3((unsigned)blocks), dim3(256), 0, s>>>(A, W, b1, b2, C, M, N, K, act);
}

static void run_lstm(hipStream_t s, const float* xin, const float* Wih, const float* Whh,
                     const float* bih, const float* bhh, float* hs, int Kin, int KinSrc) {
    const int G4 = 4 * CH;
    size_t smem = (size_t)16 * CH * 2 + (size_t)16 * CH * 4 + (size_t)16 * G4 * 4
                + (size_t)G4 * Kin * 2 + (size_t)G4 * CH * 2;   // <= 300KB of 320KB LDS
    lstm_rec_kernel<<<dim3(CNB / 16), dim3(256), smem, s>>>(
        xin, Wih, Whh, bih, bhh, hs, CNB, CT, Kin, KinSrc, CH);
}

static void run_gat(hipStream_t s, const float* xin, int Kin, const float* W,
                    const float* a_s, const float* a_d, const float* bias, int heads, int od,
                    float* hfeat, float* out, float* as_, float* ad_, unsigned* m_u,
                    float* denom, float* ex, const int* adj) {
    const int Nf = heads * od;
    run_gemm(s, xin, W, nullptr, nullptr, hfeat, CNN, Nf, Kin, 0);     // h = x @ W^T
    long nh = CNN * heads;
    gat_alpha<<<dim3((unsigned)cdivl(nh, 256)), dim3(256), 0, s>>>(hfeat, a_s, a_d, as_, ad_, heads, od);
    fill_u32<<<dim3((unsigned)cdivl(nh, 256)), dim3(256), 0, s>>>(m_u, 0u, nh);
    fill_f32<<<dim3((unsigned)cdivl(nh, 256)), dim3(256), 0, s>>>(denom, 0.f, nh);
    fill_f32<<<dim3((unsigned)cdivl(CNN * Nf, 256)), dim3(256), 0, s>>>(out, 0.f, CNN * Nf);
    long eh = CEMT * heads;
    gat_edge_max<<<dim3((unsigned)cdivl(eh, 256)), dim3(256), 0, s>>>(as_, ad_, m_u, adj, heads);
    gat_edge_exp<<<dim3((unsigned)cdivl(eh, 256)), dim3(256), 0, s>>>(as_, ad_, m_u, ex, denom, adj, heads);
    gat_aggregate<<<dim3((unsigned)cdivl(CEMT, 256)), dim3(256), 0, s>>>(hfeat, ex, denom, adj, out, heads, od);
    bias_relu<<<dim3((unsigned)cdivl(CNN * Nf, 256)), dim3(256), 0, s>>>(out, bias, CNN, Nf);
}

extern "C" void kernel_launch(void* const* d_in, const int* in_sizes, int n_in,
                              void* d_out, int out_size, void* d_ws, size_t ws_size,
                              hipStream_t stream) {
    // ---- inputs (setup_inputs order) ----
    const float* x        = (const float*)d_in[0];
    const int*   dist_adj = (const int*)d_in[1];
    const float* emb      = (const float*)d_in[3];
    const float* te_Wih0  = (const float*)d_in[4];
    const float* te_Whh0  = (const float*)d_in[5];
    const float* te_bih0  = (const float*)d_in[6];
    const float* te_bhh0  = (const float*)d_in[7];
    const float* te_Wih1  = (const float*)d_in[8];
    const float* te_Whh1  = (const float*)d_in[9];
    const float* te_bih1  = (const float*)d_in[10];
    const float* te_bhh1  = (const float*)d_in[11];
    const float* ge1_W    = (const float*)d_in[12];
    const float* ge1_as   = (const float*)d_in[13];
    const float* ge1_ad   = (const float*)d_in[14];
    const float* ge1_b    = (const float*)d_in[15];
    const float* ge2_W    = (const float*)d_in[16];
    const float* ge2_as   = (const float*)d_in[17];
    const float* ge2_ad   = (const float*)d_in[18];
    const float* ge2_b    = (const float*)d_in[19];
    const float* gefc_W   = (const float*)d_in[20];
    const float* gefc_b   = (const float*)d_in[21];
    const float* gdfc_W   = (const float*)d_in[22];
    const float* gdfc_b   = (const float*)d_in[23];
    const float* gd1_W    = (const float*)d_in[24];
    const float* gd1_as   = (const float*)d_in[25];
    const float* gd1_ad   = (const float*)d_in[26];
    const float* gd1_b    = (const float*)d_in[27];
    const float* gd2_W    = (const float*)d_in[28];
    const float* gd2_as   = (const float*)d_in[29];
    const float* gd2_ad   = (const float*)d_in[30];
    const float* gd2_b    = (const float*)d_in[31];
    const float* td_Wih0  = (const float*)d_in[32];
    const float* td_Whh0  = (const float*)d_in[33];
    const float* td_bih0  = (const float*)d_in[34];
    const float* td_bhh0  = (const float*)d_in[35];
    const float* td_Wih1  = (const float*)d_in[36];
    const float* td_Whh1  = (const float*)d_in[37];
    const float* td_bih1  = (const float*)d_in[38];
    const float* td_bhh1  = (const float*)d_in[39];
    float* out = (float*)d_out;

    // ---- workspace carving (floats): 3 big ping-pong buffers + softmax scratch
    const long BIG = CNN * (long)CH;               // 25165824 floats
    float* bufA = (float*)d_ws;
    float* bufB = bufA + BIG;
    float* bufC = bufB + BIG;
    float* sm   = bufC + BIG;
    float*    as_   = sm;                          // [nn*2]
    float*    ad_   = as_ + CNN * 2;
    unsigned* m_u   = (unsigned*)(ad_ + CNN * 2);
    float*    denom = (float*)(m_u + CNN * 2);
    float*    ex    = denom + CNN * 2;             // [EMT*2]

    // ---- temporal encoder: hin (padded K=32) -> 2-layer fused WMMA LSTM ----
    build_hin<<<dim3((unsigned)cdivl((long)CNB * CT * CK0, 256)), dim3(256), 0, stream>>>(x, emb, bufC);
    run_lstm(stream, bufC, te_Wih0, te_Whh0, te_bih0, te_bhh0, bufA, CK0, CF + CEMB);
    run_lstm(stream, bufA, te_Wih1, te_Whh1, te_bih1, te_bhh1, bufB, CH, CH);
    th_to_xf<<<dim3((unsigned)cdivl(CNN * CH, 256)), dim3(256), 0, stream>>>(bufB, bufC);

    // ---- graph encoder ----
    run_gat(stream, bufC, CH,       ge1_W, ge1_as, ge1_ad, ge1_b, 2, CHD1,
            bufA, bufB, as_, ad_, m_u, denom, ex, dist_adj);              // -> bufB [nn,128]
    run_gat(stream, bufB, 2 * CHD1, ge2_W, ge2_as, ge2_ad, ge2_b, 1, CHD1,
            bufA, bufC, as_, ad_, m_u, denom, ex, dist_adj);              // -> bufC [nn,64]
    run_gemm(stream, bufC, gefc_W, gefc_b, nullptr, bufA, CNN, CHD2, CHD1, 1);   // z [nn,32]

    // ---- graph decoder ----
    run_gemm(stream, bufA, gdfc_W, gdfc_b, nullptr, bufB, CNN, CHD1, CHD2, 1);   // [nn,64]
    run_gat(stream, bufB, CHD1,     gd1_W, gd1_as, gd1_ad, gd1_b, 2, CHD1,
            bufA, bufC, as_, ad_, m_u, denom, ex, dist_adj);              // -> bufC [nn,128]
    run_gat(stream, bufC, 2 * CHD1, gd2_W, gd2_as, gd2_ad, gd2_b, 1, CH,
            bufA, bufB, as_, ad_, m_u, denom, ex, dist_adj);              // y -> bufB [nn,128]

    // ---- temporal decoder (H=10): WMMA input-gate precompute + scalar scan
    y_to_yd<<<dim3((unsigned)cdivl(CNN * CH, 256)), dim3(256), 0, stream>>>(bufB, bufC);
    run_gemm(stream, bufC, td_Wih0, td_bih0, td_bhh0, bufA, CNN, 4 * CF, CH, 0); // gates [nn,40]
    lstm_small_l0<<<dim3(CNB / 256), dim3(256), 0, stream>>>(bufA, td_Whh0, bufB);
    lstm_small_l1_sig<<<dim3(CNB / 256), dim3(256), 0, stream>>>(
        bufB, td_Wih1, td_Whh1, td_bih1, td_bhh1, out);
}
